// BasicGNN_25950192402597
// MI455X (gfx1250) — matrix operand order repaired and checked by reading, not compiled, not run on hardware
//
#include <hip/hip_runtime.h>
#include <hip/hip_bf16.h>

// ---------------------------------------------------------------------------
// 2-layer GCN for MI455X (gfx1250, wave32).
//
// Perf model (MI455X, 23.3 TB/s HBM, 192 MB L2):
//   GEMM1 x@W1: 1.64 GFLOP but 51.2 MB of x reads -> pure memory bound.
//     -> v_wmma_f32_16x16x32_f16; A via unconditional float4 (b128) loads with
//        row clamping (no exec-mask branches); B staged in LDS pre-swizzled to
//        fragment order so each fragment is one 32B ds_load.
//   Edge aggregation: h (25.6MB) and agg (25.6MB) both L2-resident; per-edge
//     traffic ~512B served by L2; HBM only streams the 12.8MB edge list.
//     -> 16 lanes/edge, float4 gather + 4x global_atomic_add_f32 scatter.
//   Layer 2 (64->2) fused into a per-node epilogue kernel.
// ---------------------------------------------------------------------------

typedef _Float16 v16h __attribute__((ext_vector_type(16)));
typedef float    v8f  __attribute__((ext_vector_type(8)));

#define IN_F  128
#define HID_F 64
#define OUT_F 2
#define BN_EPS 1e-5f

__device__ __forceinline__ float fixf(float v) {
    // nan_to_num(nan=0, posinf=0, neginf=0): branchless cmp + cndmask
    return (v == v && fabsf(v) <= 3.402823466e38f) ? v : 0.0f;
}

// K0: deg = 1 (self loop), agg1 = 0
__global__ __launch_bounds__(256) void k_init(float* __restrict__ deg,
                                              float* __restrict__ agg1, int N) {
    int i = blockIdx.x * blockDim.x + threadIdx.x;
    if (i < N * HID_F) agg1[i] = 0.0f;
    if (i < N)         deg[i]  = 1.0f;
}

// K1: deg[dst] += 1  (float atomics are exact here: E=1.6M << 2^24)
__global__ __launch_bounds__(256) void k_degree(const int* __restrict__ dst,
                                                float* __restrict__ deg, int E) {
    int e = blockIdx.x * blockDim.x + threadIdx.x;
    if (e < E) atomicAdd(&deg[dst[e]], 1.0f);
}

// K2: dinv = rsqrt(deg), in place
__global__ __launch_bounds__(256) void k_rsqrt(float* __restrict__ deg, int N) {
    int n = blockIdx.x * blockDim.x + threadIdx.x;
    if (n < N) deg[n] = rsqrtf(deg[n]);
}

// K3: hs = (nan_to_num(x) @ W1) * dinv[row]   via v_wmma_f32_16x16x32_f16.
// Block = 256 threads = 8 waves; each wave owns a 16-row strip and produces
// the full 16x64 output tile (4 WMMA n-tiles), looping K in 4 chunks of 32.
//
// A-fragment (16-bit A 16x32, wave32): lane L -> row M = L&15, half = L>>4;
//   elements 0..7  = x[M, kb*32 + half*8 + 0..7]
//   elements 8..15 = x[M, kb*32 + 16 + half*8 + 0..7]   (two 32B chunks)
// B-fragment (32x16): lane N-col = L&15, element e -> K = kb*32 + half*8 + (e&7) + (e>>3)*16.
// LDS holds W1 pre-swizzled so each (kb,nt) fragment's per-lane 16 halves are
// contiguous: wlds[(kb*4+nt)*512 + lane*16 + e]  -> one 32B v16h load.
__global__ __launch_bounds__(256)
void k_gemm1_wmma(const float* __restrict__ x, const float* __restrict__ W1,
                  const float* __restrict__ dinv, float* __restrict__ hs, int N) {
    __shared__ __align__(32) _Float16 wlds[IN_F * HID_F];   // 16 KB
    const int tid = threadIdx.x;
    // Stage W1 -> LDS in fragment order.
    for (int i = tid; i < IN_F * HID_F; i += 256) {
        int frag = i >> 9;            // 0..15 = kb*4 + nt
        int ln   = (i >> 4) & 31;     // lane within fragment
        int e    = i & 15;            // element within lane
        int kb   = frag >> 2;
        int nt   = frag & 3;
        int k    = kb * 32 + (ln >> 4) * 8 + (e & 7) + (e >> 3) * 16;
        int n    = nt * 16 + (ln & 15);
        wlds[i] = (_Float16)W1[k * HID_F + n];
    }
    __syncthreads();

    const int lane  = tid & 31;
    const int wave  = tid >> 5;
    const int half  = lane >> 4;
    const int ncol  = lane & 15;
    const int mbase = blockIdx.x * 128 + wave * 16;
    const int row   = mbase + ncol;                 // this lane's A row
    // Clamp instead of predicating: an out-of-range A row only corrupts its
    // own D row, which is never stored. Keeps loads unconditional (b128).
    const int rowc  = row < N ? row : N - 1;
    const float4* xr4 = (const float4*)(x + (size_t)rowc * IN_F);
    const _Float16* wl = wlds + (lane << 4);

    v8f acc[4];
    acc[0] = (v8f)0.0f; acc[1] = (v8f)0.0f; acc[2] = (v8f)0.0f; acc[3] = (v8f)0.0f;

#pragma unroll
    for (int kb = 0; kb < IN_F / 32; ++kb) {
        // A: 4 unconditional float4 loads, nan_to_num, cvt to f16.
        const int c = kb * 8 + half * 2;
        float4 c0 = xr4[c];
        float4 c1 = xr4[c + 1];
        float4 c2 = xr4[c + 4];
        float4 c3 = xr4[c + 5];
        v16h a;
        a[0]  = (_Float16)fixf(c0.x); a[1]  = (_Float16)fixf(c0.y);
        a[2]  = (_Float16)fixf(c0.z); a[3]  = (_Float16)fixf(c0.w);
        a[4]  = (_Float16)fixf(c1.x); a[5]  = (_Float16)fixf(c1.y);
        a[6]  = (_Float16)fixf(c1.z); a[7]  = (_Float16)fixf(c1.w);
        a[8]  = (_Float16)fixf(c2.x); a[9]  = (_Float16)fixf(c2.y);
        a[10] = (_Float16)fixf(c2.z); a[11] = (_Float16)fixf(c2.w);
        a[12] = (_Float16)fixf(c3.x); a[13] = (_Float16)fixf(c3.y);
        a[14] = (_Float16)fixf(c3.z); a[15] = (_Float16)fixf(c3.w);
#pragma unroll
        for (int nt = 0; nt < 4; ++nt) {
            // One 32B contiguous LDS read per fragment.
            v16h b = *(const v16h*)(wl + (((kb << 2) + nt) << 9));
            acc[nt] = __builtin_amdgcn_wmma_f32_16x16x32_f16(
                false, a, false, b, (short)0, acc[nt], false, false);
        }
    }

    // D layout: lane col = L&15, VGPR r -> row = mbase + r + 8*half.
#pragma unroll
    for (int r = 0; r < 8; ++r) {
        const int rr = mbase + half * 8 + r;
        if (rr < N) {
            const float dv = dinv[rr];
            float* hrow = hs + (size_t)rr * HID_F + ncol;
#pragma unroll
            for (int nt = 0; nt < 4; ++nt)
                hrow[nt * 16] = acc[nt][r] * dv;     // pre-scaled by dinv[src]
        }
    }
}

// K4: agg1[dst] += hs[src] * dinv[dst]   (hs already carries dinv[src]).
// 16 threads per edge, float4 per thread -> coalesced 256B gather per edge.
__global__ __launch_bounds__(256)
void k_scatter1(const int* __restrict__ src, const int* __restrict__ dst,
                const float* __restrict__ dinv, const float* __restrict__ hs,
                float* __restrict__ agg1, int E) {
    int t = blockIdx.x * blockDim.x + threadIdx.x;
    int e = t >> 4;
    if (e >= E) return;
    int f = (t & 15) * 4;
    int s = src[e], d = dst[e];
    float w = dinv[d];
    const float4 v = *(const float4*)(hs + (size_t)s * HID_F + f);
    float* ap = agg1 + (size_t)d * HID_F + f;
    atomicAdd(ap + 0, v.x * w);
    atomicAdd(ap + 1, v.y * w);
    atomicAdd(ap + 2, v.z * w);
    atomicAdd(ap + 3, v.w * w);
}

// K5 (fused per-node): self-loop add + b1 + BN(eval) + ReLU + [64x2] GEMM,
// store h2s = h2*dinv (gather side of layer 2) and init out = h2*dinv^2 + b2.
__global__ __launch_bounds__(256)
void k_node_fuse(const float* __restrict__ agg1, const float* __restrict__ hs,
                 const float* __restrict__ dinv, const float* __restrict__ b1,
                 const float* __restrict__ gamma, const float* __restrict__ beta,
                 const float* __restrict__ mean, const float* __restrict__ var,
                 const float* __restrict__ W2, const float* __restrict__ b2,
                 float* __restrict__ h2s, float* __restrict__ out, int N) {
    int n = blockIdx.x * blockDim.x + threadIdx.x;
    if (n >= N) return;
    const float dv = dinv[n];
    float o0 = 0.0f, o1 = 0.0f;
    const float* ar = agg1 + (size_t)n * HID_F;
    const float* hr = hs + (size_t)n * HID_F;
#pragma unroll 4
    for (int k = 0; k < HID_F; ++k) {
        float t = ar[k] + hr[k] * dv + b1[k];          // agg + h*dinv^2 + b1
        float s = gamma[k] * rsqrtf(var[k] + BN_EPS);  // BN eval
        t = (t - mean[k]) * s + beta[k];
        t = fmaxf(t, 0.0f);                            // ReLU
        o0 += t * W2[k * 2 + 0];
        o1 += t * W2[k * 2 + 1];
    }
    h2s[n * 2 + 0] = o0 * dv;
    h2s[n * 2 + 1] = o1 * dv;
    out[n * 2 + 0] = o0 * dv * dv + b2[0];
    out[n * 2 + 1] = o1 * dv * dv + b2[1];
}

// K6: out[dst] += h2s[src] * dinv[dst]
__global__ __launch_bounds__(256)
void k_scatter2(const int* __restrict__ src, const int* __restrict__ dst,
                const float* __restrict__ dinv, const float* __restrict__ h2s,
                float* __restrict__ out, int E) {
    int e = blockIdx.x * blockDim.x + threadIdx.x;
    if (e >= E) return;
    int s = src[e], d = dst[e];
    float w = dinv[d];
    atomicAdd(&out[d * 2 + 0], h2s[s * 2 + 0] * w);
    atomicAdd(&out[d * 2 + 1], h2s[s * 2 + 1] * w);
}

static inline size_t align256(size_t x) { return (x + 255) & ~(size_t)255; }

extern "C" void kernel_launch(void* const* d_in, const int* in_sizes, int n_in,
                              void* d_out, int out_size, void* d_ws, size_t ws_size,
                              hipStream_t stream) {
    const float* x     = (const float*)d_in[0];
    const int*   edges = (const int*)d_in[1];      // [2, E]
    const float* W1    = (const float*)d_in[2];
    const float* b1    = (const float*)d_in[3];
    const float* gamma = (const float*)d_in[4];
    const float* beta  = (const float*)d_in[5];
    const float* rmean = (const float*)d_in[6];
    const float* rvar  = (const float*)d_in[7];
    const float* W2    = (const float*)d_in[8];
    const float* b2    = (const float*)d_in[9];
    float* out = (float*)d_out;

    const int N = in_sizes[0] / IN_F;
    const int E = in_sizes[1] / 2;
    const int* src = edges;
    const int* dst = edges + E;

    // Workspace layout (~52.5 MB): deg/dinv | hs | agg1 | h2s
    char* ws = (char*)d_ws;
    size_t off = 0;
    float* deg  = (float*)(ws + off); off += align256((size_t)N * sizeof(float));
    float* hs   = (float*)(ws + off); off += align256((size_t)N * HID_F * sizeof(float));
    float* agg1 = (float*)(ws + off); off += align256((size_t)N * HID_F * sizeof(float));
    float* h2s  = (float*)(ws + off); off += align256((size_t)N * OUT_F * sizeof(float));
    (void)ws_size; (void)n_in; (void)out_size;

    const int B = 256;
    k_init   <<<(N * HID_F + B - 1) / B, B, 0, stream>>>(deg, agg1, N);
    k_degree <<<(E + B - 1) / B,          B, 0, stream>>>(dst, deg, E);
    k_rsqrt  <<<(N + B - 1) / B,          B, 0, stream>>>(deg, N);
    k_gemm1_wmma<<<(N + 127) / 128,       B, 0, stream>>>(x, W1, deg, hs, N);
    k_scatter1<<<((long)E * 16 + B - 1) / B, B, 0, stream>>>(src, dst, deg, hs, agg1, E);
    k_node_fuse<<<(N + B - 1) / B,        B, 0, stream>>>(agg1, hs, deg, b1, gamma, beta,
                                                          rmean, rvar, W2, b2, h2s, out, N);
    k_scatter2<<<(E + B - 1) / B,         B, 0, stream>>>(src, dst, deg, h2s, out, E);
}
